// LSTMAutoencoder_53386443489774
// MI455X (gfx1250) — matrix operand order repaired and checked by reading, not compiled, not run on hardware
//
#include <hip/hip_runtime.h>
#include <hip/hip_bf16.h>

// Problem dims
#define B_  256
#define T_  256
#define F_  64
#define H_  256
#define G4_ 1024          // 4*H
#define KE_ 320           // F_ + H_ (encoder fused K)

typedef __attribute__((ext_vector_type(16))) _Float16 v16h;
typedef __attribute__((ext_vector_type(8)))  _Float16 v8h;
typedef __attribute__((ext_vector_type(8)))  float    v8f;

__device__ __forceinline__ float sigm(float x) {
  return 1.0f / (1.0f + __expf(-x));
}
__device__ __forceinline__ float tanha(float x) {
  float t = __expf(-2.0f * fabsf(x));
  float r = (1.0f - t) / (1.0f + t);
  return x >= 0.0f ? r : -r;
}

// A-fragment (16x32 f16): lane l<16 holds row m=l, K [kb..kb+15];
// lanes 16..31 same rows, K [kb+16..kb+31] (p = base + 8*lhalf on entry).
__device__ __forceinline__ v16h load_a_frag(const _Float16* p) {
  const int ah = (threadIdx.x >> 4) & 1;
  v8h lo = *(const v8h*)(p + ah * 8);
  v8h hi = *(const v8h*)(p + ah * 8 + 8);
  return __builtin_shufflevector(lo, hi, 0,1,2,3,4,5,6,7,8,9,10,11,12,13,14,15);
}

__device__ __forceinline__ v8f wmma16(v16h a, v16h b, v8f c) {
  return __builtin_amdgcn_wmma_f32_16x16x32_f16(false, a, false, b, (short)0, c,
                                                false, false);
}

// ---------------- prep kernels ----------------

__global__ void k_pack_enc(const float* __restrict__ Wih, const float* __restrict__ Whh,
                           const float* __restrict__ bih, const float* __restrict__ bhh,
                           _Float16* __restrict__ Wenc, float* __restrict__ bias) {
  int idx = blockIdx.x * blockDim.x + threadIdx.x;
  if (idx < G4_ * KE_) {
    int n = idx / KE_, k = idx % KE_;
    float v = (k < F_) ? Wih[n * F_ + k] : Whh[n * H_ + (k - F_)];
    Wenc[idx] = (_Float16)v;
  }
  if (idx < G4_) bias[idx] = bih[idx] + bhh[idx];
}

__global__ void k_f16cast(const float* __restrict__ src, _Float16* __restrict__ dst, int n) {
  int idx = blockIdx.x * blockDim.x + threadIdx.x;
  if (idx < n) dst[idx] = (_Float16)src[idx];
}

__global__ void k_addbias(const float* __restrict__ a, const float* __restrict__ b,
                          float* __restrict__ o, int n) {
  int idx = blockIdx.x * blockDim.x + threadIdx.x;
  if (idx < n) o[idx] = a[idx] + b[idx];
}

__global__ void k_zero(_Float16* __restrict__ h0, _Float16* __restrict__ h1,
                       float* __restrict__ c) {
  int idx = blockIdx.x * blockDim.x + threadIdx.x;
  if (idx < B_ * H_) { h0[idx] = (_Float16)0.0f; h1[idx] = (_Float16)0.0f; c[idx] = 0.0f; }
}

// ---------------- encoder recurrent step ----------------
// gates[256x1024] = [x_t | h_{t-1}] (K=320) x Wenc^T, then pointwise cell update.
// Software-pipelined: iteration kb+32's fragments are prefetched into distinct
// registers before kb's WMMAs issue, so WMMAs never wait on loadcnt==0.
__global__ void __launch_bounds__(256) k_enc_step(
    const _Float16* __restrict__ xf16, const _Float16* __restrict__ Wenc,
    const float* __restrict__ bias, const _Float16* __restrict__ h_in,
    _Float16* __restrict__ h_out, float* __restrict__ c, int t) {
  const int lane  = threadIdx.x & 31;
  const int task  = blockIdx.x * 8 + (threadIdx.x >> 5);     // 0..255
  const int mblk  = task >> 4, nblk = task & 15;
  const int mbase = mblk * 16;
  const int lhalf = (lane >> 4) & 1;
  const int arow  = mbase + (lane & 15);
  const int coln  = nblk * 16 + (lane & 15);

  const _Float16* xrow = xf16 + ((size_t)arow * T_ + t) * F_;
  const _Float16* hrow = h_in + (size_t)arow * H_;
  const _Float16* bcol = Wenc + (size_t)coln * KE_ + lhalf * 16;

#define APTR_E(kb) (((kb) < F_) ? (xrow + (kb) + lhalf * 8) : (hrow + ((kb) - F_) + lhalf * 8))

  // hoisted cell-state / bias loads (hide latency under the WMMA chain)
  float cold[8];
#pragma unroll
  for (int k = 0; k < 8; k++) cold[k] = c[(mbase + k + lhalf * 8) * H_ + coln];
  const float bi = bias[coln], bf = bias[256 + coln];
  const float bg = bias[512 + coln], bo = bias[768 + coln];

  v16h a  = load_a_frag(APTR_E(0));
  v16h b0 = *(const v16h*)(bcol);
  v16h b1 = *(const v16h*)(bcol + (size_t)1 * 256 * KE_);
  v16h b2 = *(const v16h*)(bcol + (size_t)2 * 256 * KE_);
  v16h b3 = *(const v16h*)(bcol + (size_t)3 * 256 * KE_);

  v8f acc0 = {}, acc1 = {}, acc2 = {}, acc3 = {};
#pragma unroll
  for (int kb = 0; kb < KE_; kb += 32) {
    v16h an = a, n0 = b0, n1 = b1, n2 = b2, n3 = b3;
    const int kn = kb + 32;
    if (kn < KE_) {
      an = load_a_frag(APTR_E(kn));
      const _Float16* bp = bcol + kn;
      n0 = *(const v16h*)(bp);
      n1 = *(const v16h*)(bp + (size_t)1 * 256 * KE_);
      n2 = *(const v16h*)(bp + (size_t)2 * 256 * KE_);
      n3 = *(const v16h*)(bp + (size_t)3 * 256 * KE_);
    }
    acc0 = wmma16(a, b0, acc0);
    acc1 = wmma16(a, b1, acc1);
    acc2 = wmma16(a, b2, acc2);
    acc3 = wmma16(a, b3, acc3);
    a = an; b0 = n0; b1 = n1; b2 = n2; b3 = n3;
  }
#undef APTR_E

#pragma unroll
  for (int k = 0; k < 8; k++) {
    int row = mbase + k + lhalf * 8;
    float iv = sigm(acc0[k] + bi);
    float fv = sigm(acc1[k] + bf);
    float gv = tanha(acc2[k] + bg);
    float ov = sigm(acc3[k] + bo);
    int idx = row * H_ + coln;
    float cc = fv * cold[k] + iv * gv;
    c[idx] = cc;
    h_out[idx] = (_Float16)(ov * tanha(cc));
  }
}

// ---------------- decoder one-time input-gate GEMM ----------------
// xgdec[256x1024] = h_enc x Wdec_ih^T + (b_ih+b_hh): time-invariant, computed once.
__global__ void __launch_bounds__(256) k_dec_init(
    const _Float16* __restrict__ hEnc, const _Float16* __restrict__ Wdih,
    const float* __restrict__ dbias, float* __restrict__ xgdec) {
  const int lane  = threadIdx.x & 31;
  const int task  = blockIdx.x * 8 + (threadIdx.x >> 5);     // 0..1023
  const int mblk  = task >> 6, nblk = task & 63;
  const int lhalf = (lane >> 4) & 1;
  const int arow  = mblk * 16 + (lane & 15);
  const int ncol  = nblk * 16 + (lane & 15);

  const _Float16* hrow = hEnc + (size_t)arow * H_;
  const _Float16* bp0  = Wdih + (size_t)ncol * H_ + lhalf * 16;

  v16h a = load_a_frag(hrow + lhalf * 8);
  v16h b = *(const v16h*)(bp0);
  v8f acc = {};
#pragma unroll
  for (int kb = 0; kb < H_; kb += 32) {
    v16h an = a, bn = b;
    const int kn = kb + 32;
    if (kn < H_) {
      an = load_a_frag(hrow + kn + lhalf * 8);
      bn = *(const v16h*)(bp0 + kn);
    }
    acc = wmma16(a, b, acc);
    a = an; b = bn;
  }
  const float bb = dbias[ncol];
#pragma unroll
  for (int k = 0; k < 8; k++) {
    int row = mblk * 16 + k + lhalf * 8;
    xgdec[(size_t)row * G4_ + ncol] = acc[k] + bb;
  }
}

// ---------------- decoder recurrent step (+ fused output projection) ----------------
// Tasks 0..255: gates = xgdec + h_{t-1} x Wdec_hh^T, cell update -> h_t   (t < T)
// Tasks 256..319: out[:, t-1, :] = h_{t-1} x W_out^T + b_out              (t >= 1)
__global__ void __launch_bounds__(256) k_dec_step(
    const _Float16* __restrict__ Wdhh, const float* __restrict__ xgdec,
    const _Float16* __restrict__ Wout, const float* __restrict__ bout,
    const _Float16* __restrict__ h_in, _Float16* __restrict__ h_out,
    float* __restrict__ c, float* __restrict__ out, int t) {
  const int lane  = threadIdx.x & 31;
  const int task  = blockIdx.x * 8 + (threadIdx.x >> 5);     // 0..319
  const int lhalf = (lane >> 4) & 1;

  if (task < 256) {
    if (t >= T_) return;
    const int mblk  = task >> 4, nblk = task & 15;
    const int mbase = mblk * 16;
    const int arow  = mbase + (lane & 15);
    const int coln  = nblk * 16 + (lane & 15);
    const _Float16* hrow = h_in + (size_t)arow * H_;
    const _Float16* bcol = Wdhh + (size_t)coln * H_ + lhalf * 16;

    float cold[8];
#pragma unroll
    for (int k = 0; k < 8; k++) cold[k] = c[(mbase + k + lhalf * 8) * H_ + coln];

    v16h a  = load_a_frag(hrow + lhalf * 8);
    v16h b0 = *(const v16h*)(bcol);
    v16h b1 = *(const v16h*)(bcol + (size_t)1 * 256 * H_);
    v16h b2 = *(const v16h*)(bcol + (size_t)2 * 256 * H_);
    v16h b3 = *(const v16h*)(bcol + (size_t)3 * 256 * H_);

    v8f acc0 = {}, acc1 = {}, acc2 = {}, acc3 = {};
#pragma unroll
    for (int kb = 0; kb < H_; kb += 32) {
      v16h an = a, n0 = b0, n1 = b1, n2 = b2, n3 = b3;
      const int kn = kb + 32;
      if (kn < H_) {
        an = load_a_frag(hrow + kn + lhalf * 8);
        const _Float16* bp = bcol + kn;
        n0 = *(const v16h*)(bp);
        n1 = *(const v16h*)(bp + (size_t)1 * 256 * H_);
        n2 = *(const v16h*)(bp + (size_t)2 * 256 * H_);
        n3 = *(const v16h*)(bp + (size_t)3 * 256 * H_);
      }
      acc0 = wmma16(a, b0, acc0);
      acc1 = wmma16(a, b1, acc1);
      acc2 = wmma16(a, b2, acc2);
      acc3 = wmma16(a, b3, acc3);
      a = an; b0 = n0; b1 = n1; b2 = n2; b3 = n3;
    }
#pragma unroll
    for (int k = 0; k < 8; k++) {
      int row = mbase + k + lhalf * 8;
      const float* xg = xgdec + (size_t)row * G4_ + coln;
      float iv = sigm(acc0[k] + xg[0]);
      float fv = sigm(acc1[k] + xg[256]);
      float gv = tanha(acc2[k] + xg[512]);
      float ov = sigm(acc3[k] + xg[768]);
      int idx = row * H_ + coln;
      float cc = fv * cold[k] + iv * gv;
      c[idx] = cc;
      h_out[idx] = (_Float16)(ov * tanha(cc));
    }
  } else {
    if (t == 0) return;                       // no h_{-1}
    const int idx  = task - 256;              // 0..63
    const int mblk = idx >> 2, nblk = idx & 3;
    const int arow = mblk * 16 + (lane & 15);
    const int ncol = nblk * 16 + (lane & 15);
    const _Float16* hrow = h_in + (size_t)arow * H_;
    const _Float16* bp0  = Wout + (size_t)ncol * H_ + lhalf * 16;

    v16h a = load_a_frag(hrow + lhalf * 8);
    v16h b = *(const v16h*)(bp0);
    v8f acc = {};
#pragma unroll
    for (int kb = 0; kb < H_; kb += 32) {
      v16h an = a, bn = b;
      const int kn = kb + 32;
      if (kn < H_) {
        an = load_a_frag(hrow + kn + lhalf * 8);
        bn = *(const v16h*)(bp0 + kn);
      }
      acc = wmma16(a, b, acc);
      a = an; b = bn;
    }
    const float bb = bout[ncol];
#pragma unroll
    for (int k = 0; k < 8; k++) {
      int row = mblk * 16 + k + lhalf * 8;
      out[((size_t)row * T_ + (t - 1)) * F_ + ncol] = acc[k] + bb;
    }
  }
}

extern "C" void kernel_launch(void* const* d_in, const int* in_sizes, int n_in,
                              void* d_out, int out_size, void* d_ws, size_t ws_size,
                              hipStream_t stream) {
  const float* x        = (const float*)d_in[0];
  const float* enc_W_ih = (const float*)d_in[1];
  const float* enc_W_hh = (const float*)d_in[2];
  const float* enc_b_ih = (const float*)d_in[3];
  const float* enc_b_hh = (const float*)d_in[4];
  const float* dec_W_ih = (const float*)d_in[5];
  const float* dec_W_hh = (const float*)d_in[6];
  const float* dec_b_ih = (const float*)d_in[7];
  const float* dec_b_hh = (const float*)d_in[8];
  const float* W_out    = (const float*)d_in[9];
  const float* b_out    = (const float*)d_in[10];
  float* out = (float*)d_out;

  // workspace layout (all offsets 4 KB aligned; total ~11.2 MB)
  char* ws = (char*)d_ws;
  size_t o = 0;
  _Float16* Wenc = (_Float16*)(ws + o); o += (size_t)G4_ * KE_ * 2;   // 655360
  _Float16* Wdih = (_Float16*)(ws + o); o += (size_t)G4_ * H_ * 2;    // 524288
  _Float16* Wdhh = (_Float16*)(ws + o); o += (size_t)G4_ * H_ * 2;    // 524288
  _Float16* Woutp= (_Float16*)(ws + o); o += (size_t)F_ * H_ * 2;     // 32768
  _Float16* xf16 = (_Float16*)(ws + o); o += (size_t)B_ * T_ * F_ * 2;// 8 MB
  _Float16* h0   = (_Float16*)(ws + o); o += (size_t)B_ * H_ * 2;
  _Float16* h1   = (_Float16*)(ws + o); o += (size_t)B_ * H_ * 2;
  float*    cbuf = (float*)(ws + o);    o += (size_t)B_ * H_ * 4;
  float*    ebias= (float*)(ws + o);    o += (size_t)G4_ * 4;
  float*    dbias= (float*)(ws + o);    o += (size_t)G4_ * 4;
  float*    xgdec= (float*)(ws + o);    o += (size_t)B_ * G4_ * 4;
  (void)ws_size;
  _Float16* hbuf[2] = {h0, h1};

  // ---- one-time packing / conversion (all on stream, deterministic) ----
  k_pack_enc<<<(G4_ * KE_ + 255) / 256, 256, 0, stream>>>(enc_W_ih, enc_W_hh,
                                                          enc_b_ih, enc_b_hh, Wenc, ebias);
  k_f16cast<<<(G4_ * H_ + 255) / 256, 256, 0, stream>>>(dec_W_ih, Wdih, G4_ * H_);
  k_f16cast<<<(G4_ * H_ + 255) / 256, 256, 0, stream>>>(dec_W_hh, Wdhh, G4_ * H_);
  k_f16cast<<<(F_ * H_ + 255) / 256, 256, 0, stream>>>(W_out, Woutp, F_ * H_);
  k_f16cast<<<(B_ * T_ * F_ + 255) / 256, 256, 0, stream>>>(x, xf16, B_ * T_ * F_);
  k_addbias<<<(G4_ + 255) / 256, 256, 0, stream>>>(dec_b_ih, dec_b_hh, dbias, G4_);
  k_zero<<<(B_ * H_ + 255) / 256, 256, 0, stream>>>(h0, h1, cbuf);

  // ---- encoder recurrence: step t reads h[t&1], writes h[(t+1)&1] ----
  for (int t = 0; t < T_; ++t)
    k_enc_step<<<32, 256, 0, stream>>>(xf16, Wenc, ebias,
                                       hbuf[t & 1], hbuf[(t + 1) & 1], cbuf, t);

  // final encoder hidden landed in h0 (t=255 wrote (255+1)&1 = 0)
  k_dec_init<<<128, 256, 0, stream>>>(h0, Wdih, dbias, xgdec);
  k_zero<<<(B_ * H_ + 255) / 256, 256, 0, stream>>>(h0, h1, cbuf);

  // ---- decoder recurrence with fused projection of h_{t-1}; t==T_ flushes last out ----
  for (int t = 0; t <= T_; ++t)
    k_dec_step<<<40, 256, 0, stream>>>(Wdhh, xgdec, Woutp, b_out,
                                       hbuf[t & 1], hbuf[(t + 1) & 1], cbuf, out, t);
  (void)out_size; (void)n_in; (void)in_sizes;
}
